// MultiHeadAttention_57939108823670
// MI455X (gfx1250) — matrix operand order repaired and checked by reading, not compiled, not run on hardware
//
#include <hip/hip_runtime.h>

// Problem constants (from reference)
#define B_   2
#define S_   2048
#define H_   16
#define D_   128
#define DIM_ 2048
#define E3_  (3 * H_ * D_)   // 6144

typedef __attribute__((ext_vector_type(16))) __bf16 v16bf;
typedef __attribute__((ext_vector_type(8)))  __bf16 v8bf;
typedef __attribute__((ext_vector_type(8)))  float  v8f;
typedef __attribute__((ext_vector_type(4)))  unsigned int v4u;
typedef __attribute__((ext_vector_type(8)))  int v8i;
typedef __attribute__((ext_vector_type(4)))  int v4i;

static __device__ __forceinline__ v8f vzero8f() {
  v8f z;
#pragma unroll
  for (int i = 0; i < 8; ++i) z[i] = 0.0f;
  return z;
}

static __device__ __forceinline__ v8f wmma_bf16(v16bf a, v16bf b, v8f c) {
  return __builtin_amdgcn_wmma_f32_16x16x32_bf16(
      /*neg_a=*/false, a, /*neg_b=*/false, b,
      /*c_mod=*/(short)0, c, /*reuse_a=*/false, /*reuse_b=*/false);
}

// ---- Tensor Data Mover: 2D tile load, bf16 elements, padded LDS rows.
// Tile = 32 (K, contiguous) x tile_d1 (rows), row stride `stride0` elements
// in global memory.  LDS rows padded: after every 16 DWORDs (64B = 32 bf16)
// insert 4 DWORDs (16B = 8 bf16)  ->  LDS row stride 40 elements.
static __device__ __forceinline__ void tdm_load_tile32xR(
    unsigned lds_byte_off, const void* gaddr, unsigned tile_d1, unsigned stride0) {
  unsigned long long ga = (unsigned long long)(size_t)gaddr;
  v4u g0;
  g0[0] = 1u;                                   // count=1, user mode
  g0[1] = lds_byte_off;                         // lds_addr
  g0[2] = (unsigned)ga;                         // global_addr[31:0]
  g0[3] = (unsigned)((ga >> 32) & 0x01FFFFFFu)  // global_addr[56:32]
          | 0x80000000u;                        // type=2 ("image")
  v8i g1;
  // d0: wg_mask=0, data_size=1 (2B), pad_enable=1, pad_interval=3 (16 DW),
  //     pad_amount=3 (4 DW)
  g1[0] = (int)((1u << 16) | (1u << 20) | (3u << 22) | (3u << 25));
  g1[1] = (int)((32u & 0xFFFFu) << 16);         // tensor_dim0 = 32 (lo16 @ [63:48])
  g1[2] = (int)((32u >> 16) | ((tile_d1 & 0xFFFFu) << 16)); // dim0 hi / dim1 lo
  g1[3] = (int)((tile_d1 >> 16) | (32u << 16)); // dim1 hi / tile_dim0 = 32
  g1[4] = (int)tile_d1;                         // tile_dim1 (tile_dim2 = 0)
  g1[5] = (int)stride0;                         // tensor_dim0_stride lo32
  g1[6] = 0;                                    // stride0 hi16 / stride1 lo16
  g1[7] = 0;
  v4i z4 = {0, 0, 0, 0};
  v8i z8 = {0, 0, 0, 0, 0, 0, 0, 0};
  __builtin_amdgcn_tensor_load_to_lds(g0, g1, z4, z4, z8, 0);
}

// ---- Async global->LDS 16B copy (per-lane), tracked by ASYNCcnt.
static __device__ __forceinline__ void async_ld_b128(unsigned lds_byte_off, const void* g) {
  asm volatile("global_load_async_to_lds_b128 %0, %1, off"
               :: "v"(lds_byte_off), "v"(g) : "memory");
}
static __device__ __forceinline__ void wait_asynccnt0() {
  asm volatile("s_wait_asynccnt 0x0" ::: "memory");
}

// A-operand (16x32, MxK) fragment load.  base -> element (m=0, k=0), row
// stride `stride` (elements).
static __device__ __forceinline__ v16bf ld_afrag(const __bf16* base, int stride) {
  int lane = threadIdx.x & 31;
  int m = lane & 15, half = lane >> 4;
  const __bf16* p = base + (size_t)m * stride;
  v8bf lo = *(const v8bf*)(p + half * 8);
  v8bf hi = *(const v8bf*)(p + 16 + half * 8);
  v16bf r;
#pragma unroll
  for (int i = 0; i < 8; ++i) { r[i] = lo[i]; r[i + 8] = hi[i]; }
  return r;
}

// B-operand (32x16, KxN) fragment load from an N-major (transposed) tile:
// base -> element (n=0, k=0), row stride `stride` for the N dimension.
static __device__ __forceinline__ v16bf ld_bfrag(const __bf16* base, int stride) {
  int lane = threadIdx.x & 31;
  int n = lane & 15, kh = lane >> 4;
  const __bf16* p = base + (size_t)n * stride + kh * 8;
  v8bf lo = *(const v8bf*)(p);
  v8bf hi = *(const v8bf*)(p + 16);
  v16bf r;
#pragma unroll
  for (int i = 0; i < 8; ++i) { r[i] = lo[i]; r[i + 8] = hi[i]; }
  return r;
}

// ---------------------------------------------------------------- convert
__global__ void cvt_f32_bf16(const float* __restrict__ in, __bf16* __restrict__ out, int n) {
  for (int i = blockIdx.x * blockDim.x + threadIdx.x; i < n; i += gridDim.x * blockDim.x)
    out[i] = (__bf16)in[i];
}

// ---------------------------------------------------------------- GEMM
// C[M,N] = A[M,K] * W[N,K]^T   (A, W bf16 row-major; W already N-major)
// Double-buffered K loop, tiles staged by the Tensor Data Mover.
// MODE 0: write f32 row-major to Cout
// MODE 1: scatter bf16 into q/k/v buffers laid out [B,H,S,D]
template <int MODE>
__global__ __launch_bounds__(256) void gemm_bf16(
    const __bf16* __restrict__ A, const __bf16* __restrict__ W,
    float* __restrict__ Cout,
    __bf16* __restrict__ qb, __bf16* __restrict__ kb, __bf16* __restrict__ vb,
    int M, int N, int K) {
  constexpr int LDT = 40;  // 32 + 8 pad (matches TDM pad descriptor)
  __shared__ __bf16 As[2][128 * LDT];
  __shared__ __bf16 Ws[2][128 * LDT];

  const int bn = blockIdx.x, bm = blockIdx.y;
  const int tid = threadIdx.x;
  const int w = tid >> 5, lane = tid & 31;
  const int wm = w & 3, wn = w >> 2;  // 4 waves in M, 2 in N

  v8f acc[2][4];
#pragma unroll
  for (int mt = 0; mt < 2; ++mt)
#pragma unroll
    for (int nt = 0; nt < 4; ++nt) acc[mt][nt] = vzero8f();

  const __bf16* Ab = A + (size_t)bm * 128 * K;
  const __bf16* Wb = W + (size_t)bn * 128 * K;

  if (w == 0) {  // TDM issue is wave-level (EXEC ignored)
    tdm_load_tile32xR((unsigned)(size_t)&As[0][0], Ab, 128, (unsigned)K);
    tdm_load_tile32xR((unsigned)(size_t)&Ws[0][0], Wb, 128, (unsigned)K);
  }

  int cur = 0;
  for (int k0 = 0; k0 < K; k0 += 32) {
    const bool more = (k0 + 32) < K;
    if (w == 0) {
      if (more) {
        tdm_load_tile32xR((unsigned)(size_t)&As[cur ^ 1][0], Ab + k0 + 32, 128, (unsigned)K);
        tdm_load_tile32xR((unsigned)(size_t)&Ws[cur ^ 1][0], Wb + k0 + 32, 128, (unsigned)K);
        __builtin_amdgcn_s_wait_tensorcnt(2);  // current tile's 2 loads done
      } else {
        __builtin_amdgcn_s_wait_tensorcnt(0);
      }
    }
    __syncthreads();

    v16bf af[2], wf[4];
#pragma unroll
    for (int mt = 0; mt < 2; ++mt)
      af[mt] = ld_afrag(&As[cur][(wm * 32 + mt * 16) * LDT], LDT);
#pragma unroll
    for (int nt = 0; nt < 4; ++nt)
      wf[nt] = ld_bfrag(&Ws[cur][(wn * 64 + nt * 16) * LDT], LDT);
#pragma unroll
    for (int mt = 0; mt < 2; ++mt)
#pragma unroll
      for (int nt = 0; nt < 4; ++nt) acc[mt][nt] = wmma_bf16(af[mt], wf[nt], acc[mt][nt]);

    __syncthreads();  // all waves done reading before next TDM overwrite
    cur ^= 1;
  }

  // Epilogue.  C layout: lane n = lane%16, row m = r + 8*(lane/16).
  const int hl = lane >> 4, ln = lane & 15;
  __bf16* dsts[3] = {qb, kb, vb};
#pragma unroll
  for (int mt = 0; mt < 2; ++mt)
#pragma unroll
    for (int nt = 0; nt < 4; ++nt)
#pragma unroll
      for (int r = 0; r < 8; ++r) {
        int gm = bm * 128 + wm * 32 + mt * 16 + r + 8 * hl;
        int gn = bn * 128 + wn * 64 + nt * 16 + ln;
        float val = acc[mt][nt][r];
        if constexpr (MODE == 0) {
          Cout[(size_t)gm * N + gn] = val;
        } else {
          int which = gn >> 11;            // 0:q 1:k 2:v (chunks of H*D=2048)
          int rem   = gn & 2047;
          int h = rem >> 7, d = rem & 127;
          int b = gm >> 11, s = gm & 2047; // token = b*S + s
          dsts[which][(((size_t)(b * H_ + h) * S_) + s) * D_ + d] = (__bf16)val;
        }
      }
}

// ---------------------------------------------------------------- RoPE
// In-place rotation on q (with 1/sqrt(D) scale folded in) and k, [B,H,S,D].
__global__ void rope_kernel(__bf16* __restrict__ q, __bf16* __restrict__ k, float scale) {
  int idx = blockIdx.x * blockDim.x + threadIdx.x;  // B*H*S*(D/2) threads
  int i = idx & 63;
  int s = (idx >> 6) & (S_ - 1);
  int bh = idx >> 17;
  float freq = __expf(-(float)(2 * i) * 0.0719557842f);  // ln(1e4)/128
  float c, sn;
  __sincosf((float)s * freq, &sn, &c);
  size_t base = ((size_t)bh * S_ + s) * D_;
  {
    float a = (float)q[base + i], b = (float)q[base + i + 64];
    q[base + i]      = (__bf16)((a * c - b * sn) * scale);
    q[base + i + 64] = (__bf16)((b * c + a * sn) * scale);
  }
  {
    float a = (float)k[base + i], b = (float)k[base + i + 64];
    k[base + i]      = (__bf16)(a * c - b * sn);
    k[base + i + 64] = (__bf16)(b * c + a * sn);
  }
}

// ---------------------------------------------------------------- attention
// One block: (q-block of 128 rows) x (one b,h).  8 waves, 16 q-rows per wave.
// Flash-attention over 64-key blocks, bf16 WMMA, f32 online softmax.
// K tiles staged with async global->LDS; V transposed via LDS scatter.
__global__ __launch_bounds__(256) void attn_kernel(
    const __bf16* __restrict__ Q, const __bf16* __restrict__ Kv,
    const __bf16* __restrict__ V, __bf16* __restrict__ O /*[B,S,H*D]*/) {
  constexpr int LKD = 136;  // 128 + 8 pad
  constexpr int LVT = 72;   // 64 + 8 pad
  constexpr int LP  = 72;
  __shared__ __bf16 Ks[64 * LKD];       // K tile, row-major [key][d]
  __shared__ __bf16 Vts[128 * LVT];     // V tile transposed [d][key]
  __shared__ __bf16 Ps[8 * 16 * LP];    // per-wave P tiles

  const int qblk = blockIdx.x;    // S/128
  const int bh   = blockIdx.y;    // B*H
  const int b = bh >> 4, h = bh & 15;
  const int w = threadIdx.x >> 5, lane = threadIdx.x & 31;
  const int m16 = lane & 15, hl = lane >> 4;
  const int q0 = qblk * 128;
  const int qrow = q0 + w * 16;

  // Load this wave's Q fragments straight from global (A-operand layout).
  const __bf16* Qb = Q + ((size_t)bh * S_ + qrow + m16) * D_;
  v16bf qf[4];
#pragma unroll
  for (int kc = 0; kc < 4; ++kc) {
    v8bf lo = *(const v8bf*)&Qb[kc * 32 + hl * 8];
    v8bf hi = *(const v8bf*)&Qb[kc * 32 + 16 + hl * 8];
#pragma unroll
    for (int i = 0; i < 8; ++i) { qf[kc][i] = lo[i]; qf[kc][i + 8] = hi[i]; }
  }

  float runmax[8], lsum[8];
  v8f oacc[8];
#pragma unroll
  for (int r = 0; r < 8; ++r) { runmax[r] = -1e30f; lsum[r] = 0.0f; }
#pragma unroll
  for (int dt = 0; dt < 8; ++dt) oacc[dt] = vzero8f();

  const int qhi = qrow + 15;
  const float MASKV = -3.0e38f;

  for (int kb = 0; kb < q0 + 128; kb += 64) {
    // ---- stage K (async -> LDS) and V (transposed scatter)
    {
      int r  = threadIdx.x >> 2;
      int c0 = (threadIdx.x & 3) * 32;
      const __bf16* Kp = Kv + ((size_t)bh * S_ + kb + r) * D_ + c0;
      const __bf16* Vp = V  + ((size_t)bh * S_ + kb + r) * D_ + c0;
#pragma unroll
      for (int j = 0; j < 4; ++j)
        async_ld_b128((unsigned)(size_t)&Ks[r * LKD + c0 + j * 8], Kp + j * 8);
#pragma unroll
      for (int j = 0; j < 32; ++j) Vts[(c0 + j) * LVT + r] = Vp[j];
      if (kb + 64 < q0 + 128) {  // warm L2 for the next key block
        __builtin_prefetch(Kp + (size_t)64 * D_, 0, 1);
        __builtin_prefetch(Vp + (size_t)64 * D_, 0, 1);
      }
    }
    wait_asynccnt0();
    __syncthreads();

    if (kb <= qhi) {  // wave-uniform
      // ---- S = Q * K^T  (16 x 64 tile)
      v8f sacc[4];
#pragma unroll
      for (int nt = 0; nt < 4; ++nt) sacc[nt] = vzero8f();
#pragma unroll
      for (int nt = 0; nt < 4; ++nt)
#pragma unroll
        for (int kc = 0; kc < 4; ++kc) {
          v16bf kf = ld_bfrag(&Ks[(nt * 16) * LKD + kc * 32], LKD);
          sacc[nt] = wmma_bf16(qf[kc], kf, sacc[nt]);
        }

      // ---- causal mask + online softmax (rows live in C layout)
      float p[4][8];
#pragma unroll
      for (int r = 0; r < 8; ++r) {
        int qg = qrow + r + 8 * hl;
        float mx = MASKV;
#pragma unroll
        for (int nt = 0; nt < 4; ++nt) {
          int key = kb + nt * 16 + m16;
          float sv = (key <= qg) ? sacc[nt][r] : MASKV;
          p[nt][r] = sv;
          mx = fmaxf(mx, sv);
        }
#pragma unroll
        for (int off = 1; off < 16; off <<= 1) mx = fmaxf(mx, __shfl_xor(mx, off, 32));
        float nm = fmaxf(runmax[r], mx);
        float corr = __expf(runmax[r] - nm);
        runmax[r] = nm;
        float rs = 0.0f;
#pragma unroll
        for (int nt = 0; nt < 4; ++nt) {
          float e = __expf(p[nt][r] - nm);
          p[nt][r] = e;
          rs += e;
        }
#pragma unroll
        for (int off = 1; off < 16; off <<= 1) rs += __shfl_xor(rs, off, 32);
        lsum[r] = lsum[r] * corr + rs;
#pragma unroll
        for (int dt = 0; dt < 8; ++dt) oacc[dt][r] *= corr;
      }

      // ---- P -> LDS (C layout -> A layout round trip), then O += P * V
      __bf16* Pw = &Ps[w * 16 * LP];
#pragma unroll
      for (int nt = 0; nt < 4; ++nt)
#pragma unroll
        for (int r = 0; r < 8; ++r)
          Pw[(r + 8 * hl) * LP + nt * 16 + m16] = (__bf16)p[nt][r];

#pragma unroll
      for (int dt = 0; dt < 8; ++dt)
#pragma unroll
        for (int kc2 = 0; kc2 < 2; ++kc2) {
          v16bf pf = ld_afrag(&Pw[kc2 * 32], LP);
          v16bf vf = ld_bfrag(&Vts[(dt * 16) * LVT + kc2 * 32], LVT);
          oacc[dt] = wmma_bf16(pf, vf, oacc[dt]);
        }
    }
    __syncthreads();
  }

  // ---- normalize and write [B,S,H*D] bf16
#pragma unroll
  for (int r = 0; r < 8; ++r) {
    float inv = 1.0f / lsum[r];
    int s = qrow + r + 8 * hl;
    size_t rowbase = ((size_t)b * S_ + s) * (H_ * D_) + h * D_;
#pragma unroll
    for (int dt = 0; dt < 8; ++dt)
      O[rowbase + dt * 16 + m16] = (__bf16)(oacc[dt][r] * inv);
  }
}

// ---------------------------------------------------------------- launch
extern "C" void kernel_launch(void* const* d_in, const int* in_sizes, int n_in,
                              void* d_out, int out_size, void* d_ws, size_t ws_size,
                              hipStream_t stream) {
  const float* x     = (const float*)d_in[0];   // [B,S,DIM]
  const float* qkv_w = (const float*)d_in[1];   // [3*H*D, DIM]
  const float* out_w = (const float*)d_in[2];   // [DIM, DIM]
  float* out = (float*)d_out;                   // [B,S,DIM] f32

  char* ws = (char*)d_ws;
  size_t off = 0;
  auto alloc = [&](size_t bytes) {
    char* p = ws + off;
    off += (bytes + 255) & ~(size_t)255;
    return p;
  };
  const size_t NTOK = (size_t)B_ * S_;          // 4096
  __bf16* xb   = (__bf16*)alloc(NTOK * DIM_ * 2);
  __bf16* wqb  = (__bf16*)alloc((size_t)E3_ * DIM_ * 2);
  __bf16* wob  = (__bf16*)alloc((size_t)DIM_ * DIM_ * 2);
  __bf16* qb   = (__bf16*)alloc(NTOK * DIM_ * 2);
  __bf16* kbuf = (__bf16*)alloc(NTOK * DIM_ * 2);
  __bf16* vbuf = (__bf16*)alloc(NTOK * DIM_ * 2);
  __bf16* ao   = (__bf16*)alloc(NTOK * DIM_ * 2);
  (void)ws_size; (void)in_sizes; (void)n_in; (void)out_size;

  // 1. convert inputs to bf16
  cvt_f32_bf16<<<2048, 256, 0, stream>>>(x, xb, (int)(NTOK * DIM_));
  cvt_f32_bf16<<<2048, 256, 0, stream>>>(qkv_w, wqb, E3_ * DIM_);
  cvt_f32_bf16<<<2048, 256, 0, stream>>>(out_w, wob, DIM_ * DIM_);

  // 2. QKV projection -> scatter into q/k/v [B,H,S,D]
  gemm_bf16<1><<<dim3(E3_ / 128, NTOK / 128), 256, 0, stream>>>(
      xb, wqb, nullptr, qb, kbuf, vbuf, (int)NTOK, E3_, DIM_);

  // 3. RoPE in place on q (scaled by 1/sqrt(D)) and k
  {
    int total = B_ * H_ * S_ * (D_ / 2);
    rope_kernel<<<total / 256, 256, 0, stream>>>(qb, kbuf, 0.08838834764831845f);
  }

  // 4. causal flash attention -> ao [B,S,H*D] bf16
  attn_kernel<<<dim3(S_ / 128, B_ * H_), 256, 0, stream>>>(qb, kbuf, vbuf, ao);

  // 5. output projection -> f32 d_out
  gemm_bf16<0><<<dim3(DIM_ / 128, NTOK / 128), 256, 0, stream>>>(
      ao, wob, out, nullptr, nullptr, nullptr, (int)NTOK, DIM_, DIM_);
}
